// CNNMRF_58909771432230
// MI455X (gfx1250) — matrix operand order repaired and checked by reading, not compiled, not run on hardware
//
#include <hip/hip_runtime.h>

// MRF loss, fused for MI455X (gfx1250, wave32, WMMA):
//   1) build L2-normalized f16 patch matrices Ch, Sh (N x D), D = 128*9 = 1152
//   2) fused f16-WMMA GEMM (Ch x Sh^T) with streaming row-argmax (no 340MB dist)
//   3) gather matched style patches, fold, divisor, MSE -> scalar
typedef __attribute__((ext_vector_type(16))) _Float16 v16h;
typedef __attribute__((ext_vector_type(8)))  _Float16 v8h;
typedef __attribute__((ext_vector_type(8)))  float    v8f;

#define Hh 96
#define Ww 96
#define Cc 128
#define NP (Hh*Ww)        // 9216 patches
#define Dd (Cc*9)         // 1152
#define KBLK (Dd/32)      // 36 K-blocks of 32
#define TILES (NP/16)     // 576 16-row tiles

// ---------------- Phase 1: normalized f16 patch matrices --------------------
// One block (256 threads) per patch; d-index layout: d = (ki*3+kj)*128 + c.
__global__ void build_patches(const float* __restrict__ content,
                              const float* __restrict__ style,
                              _Float16* __restrict__ Ch,
                              _Float16* __restrict__ Sh) {
    int b = blockIdx.x;
    const float* src;
    _Float16* dst;
    int n;
    if (b < NP) { src = content; dst = Ch; n = b; }
    else        { src = style;   dst = Sh; n = b - NP; }
    int py = n / Ww, px = n % Ww;
    int t = threadIdx.x;
    float vals[5];
    float ss = 0.f;
#pragma unroll
    for (int it = 0; it < 5; ++it) {
        int d = t + it * 256;
        float v = 0.f;
        if (d < Dd) {
            int k = d >> 7, c = d & 127;
            int ki = k / 3, kj = k % 3;
            int y = py + ki - 1, x = px + kj - 1;
            if (y >= 0 && y < Hh && x >= 0 && x < Ww)
                v = src[(size_t)c * NP + y * Ww + x];
        }
        vals[it] = v;
        ss += v * v;
    }
    __shared__ float red[256];
    red[t] = ss;
    __syncthreads();
    for (int s = 128; s > 0; s >>= 1) {
        if (t < s) red[t] += red[t + s];
        __syncthreads();
    }
    float inv = 1.0f / fmaxf(sqrtf(red[0]), 1e-12f);
#pragma unroll
    for (int it = 0; it < 5; ++it) {
        int d = t + it * 256;
        if (d < Dd) dst[(size_t)n * Dd + d] = (_Float16)(vals[it] * inv);
    }
}

// ---------------- Phase 2: WMMA GEMM + fused row-argmax ---------------------
// One wave per 16-row content tile; 4 style tiles (4 accumulators) per pass.
__global__ void __launch_bounds__(256)
wmma_argmax(const _Float16* __restrict__ Ch,
            const _Float16* __restrict__ Sh,
            int* __restrict__ bestIdx) {
    int lane = threadIdx.x & 31;
    int wave = threadIdx.x >> 5;
    int tr = blockIdx.x * 8 + wave;     // 0..575
    int rowBase = tr * 16;
    int l15 = lane & 15;
    int hi  = lane >> 4;                // 0 or 1

    // A fragment base: row = l15; lane-half selects K {0..15} vs {16..31}
    const _Float16* aBase = Ch + (size_t)(rowBase + l15) * Dd + (hi ? 16 : 0);

    float bVal[8];
    int   bIdx[8];
#pragma unroll
    for (int v = 0; v < 8; ++v) { bVal[v] = -1e30f; bIdx[v] = 0; }

    for (int g = 0; g < TILES / 4; ++g) {
        v8f acc0 = {}, acc1 = {}, acc2 = {}, acc3 = {};
        int col0 = g * 64 + l15;
        // B fragment: col = l15; lane-half selects K {0..15} vs {16..31}
        const _Float16* p0 = Sh + (size_t)(col0     ) * Dd + (hi ? 16 : 0);
        const _Float16* p1 = Sh + (size_t)(col0 + 16) * Dd + (hi ? 16 : 0);
        const _Float16* p2 = Sh + (size_t)(col0 + 32) * Dd + (hi ? 16 : 0);
        const _Float16* p3 = Sh + (size_t)(col0 + 48) * Dd + (hi ? 16 : 0);
        for (int kb = 0; kb < KBLK; ++kb) {
            int k0 = kb * 32;
            v8h alo = *(const v8h*)(aBase + k0);
            v8h ahi = *(const v8h*)(aBase + k0 + 8);
            v16h a = __builtin_shufflevector(alo, ahi,
                0,1,2,3,4,5,6,7,8,9,10,11,12,13,14,15);

            v8h q0 = *(const v8h*)(p0 + k0);
            v8h q1 = *(const v8h*)(p0 + k0 + 8);
            v16h bm = __builtin_shufflevector(q0, q1,
                0,1,2,3,4,5,6,7,8,9,10,11,12,13,14,15);
            acc0 = __builtin_amdgcn_wmma_f32_16x16x32_f16(
                false, a, false, bm, (short)0, acc0, false, false);

            q0 = *(const v8h*)(p1 + k0);
            q1 = *(const v8h*)(p1 + k0 + 8);
            bm = __builtin_shufflevector(q0, q1,
                0,1,2,3,4,5,6,7,8,9,10,11,12,13,14,15);
            acc1 = __builtin_amdgcn_wmma_f32_16x16x32_f16(
                false, a, false, bm, (short)0, acc1, false, false);

            q0 = *(const v8h*)(p2 + k0);
            q1 = *(const v8h*)(p2 + k0 + 8);
            bm = __builtin_shufflevector(q0, q1,
                0,1,2,3,4,5,6,7,8,9,10,11,12,13,14,15);
            acc2 = __builtin_amdgcn_wmma_f32_16x16x32_f16(
                false, a, false, bm, (short)0, acc2, false, false);

            q0 = *(const v8h*)(p3 + k0);
            q1 = *(const v8h*)(p3 + k0 + 8);
            bm = __builtin_shufflevector(q0, q1,
                0,1,2,3,4,5,6,7,8,9,10,11,12,13,14,15);
            acc3 = __builtin_amdgcn_wmma_f32_16x16x32_f16(
                false, a, false, bm, (short)0, acc3, false, false);
        }
        // D layout: VGPR v, this lane holds (M = v + 8*hi, N = l15).
        // Update running best over style index; '>' keeps first max.
#pragma unroll
        for (int v = 0; v < 8; ++v) {
            float x;
            x = acc0[v]; if (x > bVal[v]) { bVal[v] = x; bIdx[v] = col0;      }
            x = acc1[v]; if (x > bVal[v]) { bVal[v] = x; bIdx[v] = col0 + 16; }
            x = acc2[v]; if (x > bVal[v]) { bVal[v] = x; bIdx[v] = col0 + 32; }
            x = acc3[v]; if (x > bVal[v]) { bVal[v] = x; bIdx[v] = col0 + 48; }
        }
    }
    // Cross-lane argmax within each 16-lane half (masks < 16 stay in-half).
#pragma unroll
    for (int v = 0; v < 8; ++v) {
        float val = bVal[v];
        int idx = bIdx[v];
#pragma unroll
        for (int m = 1; m < 16; m <<= 1) {
            float ov = __shfl_xor(val, m, 32);
            int   oi = __shfl_xor(idx, m, 32);
            if (ov > val || (ov == val && oi < idx)) { val = ov; idx = oi; }
        }
        if (l15 == 0) bestIdx[rowBase + v + 8 * hi] = idx;
    }
}

// ---------------- Phase 3: fold matched patches + MSE -----------------------
// Thread per (c,y,x). recon[c,y,x] = sum over covering patches of the matched
// style value; divide by overlap count + 1e-8; accumulate squared error.
__global__ void mrf_loss(const float* __restrict__ content,
                         const float* __restrict__ style,
                         const int* __restrict__ bestIdx,
                         float* __restrict__ partial) {
    int gid = blockIdx.x * 256 + threadIdx.x;   // exactly C*H*W threads
    int c = gid / NP;
    int pix = gid - c * NP;
    int y = pix / Ww, x = pix % Ww;
    float sum = 0.f;
    int cnt = 0;
#pragma unroll
    for (int ki = 0; ki < 3; ++ki) {
        int py = y + 1 - ki;
        if (py < 0 || py >= Hh) continue;
#pragma unroll
        for (int kj = 0; kj < 3; ++kj) {
            int px = x + 1 - kj;
            if (px < 0 || px >= Ww) continue;
            int m = bestIdx[py * Ww + px];
            int my = m / Ww, mx = m % Ww;
            int sy = my + ki - 1, sx = mx + kj - 1;
            float v = 0.f;
            if (sy >= 0 && sy < Hh && sx >= 0 && sx < Ww)
                v = style[(size_t)c * NP + sy * Ww + sx];
            sum += v;
            cnt += 1;
        }
    }
    float recon = sum / ((float)cnt + 1e-8f);
    float d = content[gid] - recon;
    float sq = d * d;
    __shared__ float red[256];
    red[threadIdx.x] = sq;
    __syncthreads();
    for (int s = 128; s > 0; s >>= 1) {
        if (threadIdx.x < s) red[threadIdx.x] += red[threadIdx.x + s];
        __syncthreads();
    }
    if (threadIdx.x == 0) partial[blockIdx.x] = red[0];
}

__global__ void final_reduce(const float* __restrict__ partial,
                             float* __restrict__ out, int nblocks) {
    __shared__ float red[256];
    float s = 0.f;
    for (int i = threadIdx.x; i < nblocks; i += 256) s += partial[i];
    red[threadIdx.x] = s;
    __syncthreads();
    for (int k = 128; k > 0; k >>= 1) {
        if (threadIdx.x < k) red[threadIdx.x] += red[threadIdx.x + k];
        __syncthreads();
    }
    if (threadIdx.x == 0) out[0] = red[0] / (float)(Cc * NP);
}

extern "C" void kernel_launch(void* const* d_in, const int* in_sizes, int n_in,
                              void* d_out, int out_size, void* d_ws, size_t ws_size,
                              hipStream_t stream) {
    const float* content = (const float*)d_in[0];  // (1,128,96,96) f32
    const float* style   = (const float*)d_in[1];  // (1,128,96,96) f32

    // Workspace layout (~40.6 MiB):
    //   Ch:      NP*Dd f16  (21,233,664 B)
    //   Sh:      NP*Dd f16  (21,233,664 B)
    //   bestIdx: NP int     (36,864 B)
    //   partial: 4608 f32   (18,432 B)
    char* ws = (char*)d_ws;
    const size_t matBytes = (size_t)NP * Dd * sizeof(_Float16);
    _Float16* Ch = (_Float16*)ws;
    _Float16* Sh = (_Float16*)(ws + matBytes);
    int* bidx    = (int*)(ws + 2 * matBytes);
    float* partial = (float*)(ws + 2 * matBytes + (size_t)NP * sizeof(int));

    build_patches<<<2 * NP, 256, 0, stream>>>(content, style, Ch, Sh);
    wmma_argmax<<<TILES / 8, 256, 0, stream>>>(Ch, Sh, bidx);
    const int nblk = (Cc * NP) / 256;   // 4608
    mrf_loss<<<nblk, 256, 0, stream>>>(content, style, bidx, partial);
    final_reduce<<<1, 256, 0, stream>>>(partial, (float*)d_out, nblk);
}
